// GAT_P_43533788512517
// MI455X (gfx1250) — compile-verified, matrix operand order checked by reading, not run on hardware
//
#include <hip/hip_runtime.h>
#include <math.h>

typedef __attribute__((ext_vector_type(16))) _Float16 v16h;
typedef __attribute__((ext_vector_type(8)))  _Float16 v8h;
typedef __attribute__((ext_vector_type(8)))  float    v8f;

#define TPB 256

// ---------------------------------------------------------------- utilities

__global__ void fill_f32_k(float* __restrict__ p, float v, long n) {
    long t = (long)blockIdx.x * blockDim.x + threadIdx.x;
    if (t < n) p[t] = v;
}

__global__ void cvt_f16_k(const float* __restrict__ in, _Float16* __restrict__ out, long n) {
    long t = (long)blockIdx.x * blockDim.x + threadIdx.x;
    if (t < n) out[t] = (_Float16)in[t];
}

// W [K x N] f32 row-major  ->  Wt [N x K] f16 row-major (transposed)
__global__ void transpose_f16_k(const float* __restrict__ W, _Float16* __restrict__ Wt,
                                int K, int N) {
    int t = blockIdx.x * blockDim.x + threadIdx.x;
    if (t >= K * N) return;
    int n = t % N, k = t / N;
    Wt[n * K + k] = (_Float16)W[k * N + n];
}

__device__ __forceinline__ void atomicMaxF(float* addr, float val) {
    // Mixed-sign float atomic max: positive floats order as ints,
    // negative floats order reversed as uints.
    if (val >= 0.0f) atomicMax((int*)addr, __float_as_int(val));
    else             atomicMin((unsigned int*)addr, __float_as_uint(val));
}

__device__ __forceinline__ void edge_sd(const int* __restrict__ ei, int E, int e,
                                        int& s, int& d) {
    if (e < E) { s = ei[e]; d = ei[E + e]; }
    else       { s = e - E; d = e - E; }       // self loops appended
}

// ---------------------------------------------------------------- WMMA GEMM
// ISA 16-bit fragment layout: lanes 0-15 hold K {k0..k0+7, k0+16..k0+23} of
// row/col = lane; lanes 16-31 hold K {k0+8..k0+15, k0+24..k0+31}.
// Both runs are 8 contiguous halfs -> two 128-bit loads.
__device__ __forceinline__ v16h load_frag16(const _Float16* __restrict__ p) {
    v8h lo = *(const v8h*)(p);        // slots 0-7
    v8h hi = *(const v8h*)(p + 16);   // slots 8-15
    v16h r;
    #pragma unroll
    for (int i = 0; i < 8; ++i) { r[i] = lo[i]; r[8 + i] = hi[i]; }
    return r;
}

// C[M x 512] = Ah[M x 512](f16) * B[512 x 512]; Bt = B^T as [N=512 x K=512] f16.
// One wave computes a 16x64 strip (4 accumulators, A fragment reused 4x).
__global__ void __launch_bounds__(TPB)
gemm_wmma_f16_k(const _Float16* __restrict__ Ah, const _Float16* __restrict__ Bt,
                float* __restrict__ C, int M) {
    const int K = 512, Nc = 512, tilesN4 = Nc / 64;   // 8 strips of 64 cols
    int wave = threadIdx.x >> 5;
    int lane = threadIdx.x & 31;
    int tile = blockIdx.x * 8 + wave;
    int total = (M / 16) * tilesN4;
    if (tile >= total) return;                 // wave-uniform -> EXEC stays all-1s
    int tm = tile / tilesN4, tn4 = tile % tilesN4;
    int mrow = lane & 15;                      // A row / B col within tile
    int hi   = lane >> 4;                      // lane half selects K sub-group
    int koff = hi * 8;

    const _Float16* arow = Ah + (long)(tm * 16 + mrow) * K + koff;
    const _Float16* br0 = Bt + (long)(tn4 * 64 +  0 + mrow) * K + koff;
    const _Float16* br1 = Bt + (long)(tn4 * 64 + 16 + mrow) * K + koff;
    const _Float16* br2 = Bt + (long)(tn4 * 64 + 32 + mrow) * K + koff;
    const _Float16* br3 = Bt + (long)(tn4 * 64 + 48 + mrow) * K + koff;

    v8f acc0 = {}, acc1 = {}, acc2 = {}, acc3 = {};
    #pragma unroll 2
    for (int k0 = 0; k0 < K; k0 += 32) {
        v16h a  = load_frag16(arow + k0);
        v16h b0 = load_frag16(br0 + k0);
        v16h b1 = load_frag16(br1 + k0);
        v16h b2 = load_frag16(br2 + k0);
        v16h b3 = load_frag16(br3 + k0);
        acc0 = __builtin_amdgcn_wmma_f32_16x16x32_f16(false, a, false, b0, (short)0, acc0, false, false);
        acc1 = __builtin_amdgcn_wmma_f32_16x16x32_f16(false, a, false, b1, (short)0, acc1, false, false);
        acc2 = __builtin_amdgcn_wmma_f32_16x16x32_f16(false, a, false, b2, (short)0, acc2, false, false);
        acc3 = __builtin_amdgcn_wmma_f32_16x16x32_f16(false, a, false, b3, (short)0, acc3, false, false);
    }
    // D layout: lane -> col = mrow, rows v + 8*hi
    float* crow = C + (long)(tm * 16 + 8 * hi) * Nc + tn4 * 64 + mrow;
    #pragma unroll
    for (int v = 0; v < 8; ++v) {
        crow[(long)v * Nc +  0] = acc0[v];
        crow[(long)v * Nc + 16] = acc1[v];
        crow[(long)v * Nc + 32] = acc2[v];
        crow[(long)v * Nc + 48] = acc3[v];
    }
}

// ---------------------------------------------------------------- attention
// out[n,h] = sum_c xp[n,h,c] * att[h,c]
__global__ void att_score_k(const float* __restrict__ xp, const float* __restrict__ att,
                            float* __restrict__ out, int N, int H, int Cd) {
    int t = blockIdx.x * blockDim.x + threadIdx.x;
    if (t >= N * H) return;
    int h = t % H, n = t / H;
    const float* xr = xp + (long)n * H * Cd + h * Cd;
    const float* ar = att + h * Cd;
    float s = 0.f;
    for (int c = 0; c < Cd; ++c) s += xr[c] * ar[c];
    out[t] = s;
}

__global__ void edge_max_k(const int* __restrict__ ei, int E, int N, int H,
                           const float* __restrict__ asrc, const float* __restrict__ adst,
                           float* __restrict__ m) {
    int t = blockIdx.x * blockDim.x + threadIdx.x;
    if (t >= (E + N) * H) return;
    int h = t % H, e = t / H;
    int s, d; edge_sd(ei, E, e, s, d);
    float v = asrc[s * H + h] + adst[d * H + h];
    v = v > 0.f ? v : 0.2f * v;                       // leaky_relu 0.2
    atomicMaxF(&m[d * H + h], v);
}

__global__ void edge_exp_k(const int* __restrict__ ei, int E, int N, int H,
                           const float* __restrict__ asrc, const float* __restrict__ adst,
                           const float* __restrict__ m, float* __restrict__ ex,
                           float* __restrict__ denom) {
    int t = blockIdx.x * blockDim.x + threadIdx.x;
    if (t >= (E + N) * H) return;
    int h = t % H, e = t / H;
    int s, d; edge_sd(ei, E, e, s, d);
    float v = asrc[s * H + h] + adst[d * H + h];
    v = v > 0.f ? v : 0.2f * v;
    float ev = expf(v - m[d * H + h]);
    ex[t] = ev;
    atomicAdd(&denom[d * H + h], ev);
}

// out[dst, h, c] += alpha * xp[src, h, c]; thread handles 8 channels
__global__ void agg_k(const int* __restrict__ ei, int E, int N, int H, int Cd,
                      const float* __restrict__ ex, const float* __restrict__ denom,
                      const float* __restrict__ xp, float* __restrict__ out) {
    int cpt = Cd / 8;
    long t = (long)blockIdx.x * blockDim.x + threadIdx.x;
    long total = (long)(E + N) * H * cpt;
    if (t >= total) return;
    int chunk = (int)(t % cpt);
    long r = t / cpt;
    int h = (int)(r % H);
    int e = (int)(r / H);
    int s, d; edge_sd(ei, E, e, s, d);
    float w = ex[e * H + h] / (denom[d * H + h] + 1e-16f);
    const float* xs = xp  + (long)s * H * Cd + h * Cd + chunk * 8;
    float*       od = out + (long)d * H * Cd + h * Cd + chunk * 8;
    #pragma unroll
    for (int i = 0; i < 8; ++i) atomicAdd(&od[i], w * xs[i]);
}

// h_f16 = (f16) elu(agg + b1)   (f16 copy feeds the second WMMA GEMM)
__global__ void elu_bias_f16_k(const float* __restrict__ h, const float* __restrict__ b,
                               _Float16* __restrict__ out, long total, int HC) {
    long t = (long)blockIdx.x * blockDim.x + threadIdx.x;
    if (t >= total) return;
    float v = h[t] + b[t % HC];
    v = v > 0.f ? v : expm1f(v);
    out[t] = (_Float16)v;
}

__global__ void mean_bias_k(const float* __restrict__ agg, const float* __restrict__ b,
                            float* __restrict__ out, int N, int H, int Cd) {
    int t = blockIdx.x * blockDim.x + threadIdx.x;
    if (t >= N * Cd) return;
    int c = t % Cd, n = t / Cd;
    float s = 0.f;
    for (int h = 0; h < H; ++h) s += agg[(long)n * H * Cd + h * Cd + c];
    out[t] = s / (float)H + b[c];
}

__global__ void ce_k(const float* __restrict__ logits, const int* __restrict__ y,
                     const unsigned char* __restrict__ mask, float* __restrict__ acc,
                     int N, int Cd) {
    int n = blockIdx.x * blockDim.x + threadIdx.x;
    if (n >= N) return;
    const float* row = logits + (long)n * Cd;
    float mx = -1e30f;
    for (int c = 0; c < Cd; ++c) mx = fmaxf(mx, row[c]);
    float s = 0.f;
    for (int c = 0; c < Cd; ++c) s += expf(row[c] - mx);
    float ce = (mx + logf(s)) - row[y[n]];
    float w = mask[n] ? 1.0f : 0.0f;
    atomicAdd(&acc[0], w * ce);
    atomicAdd(&acc[1], w);
}

__global__ void finalize_k(const float* __restrict__ acc, float* __restrict__ out) {
    out[0] = acc[0] / acc[1];
}

// ---------------------------------------------------------------- host

static inline int gdiv(long n, int b) { return (int)((n + b - 1) / b); }

extern "C" void kernel_launch(void* const* d_in, const int* in_sizes, int n_in,
                              void* d_out, int out_size, void* d_ws, size_t ws_size,
                              hipStream_t stream) {
    const float* x        = (const float*)d_in[0];
    const int*   ei       = (const int*)  d_in[1];
    const int*   y        = (const int*)  d_in[2];
    const unsigned char* mask = (const unsigned char*)d_in[3];
    const float* W1       = (const float*)d_in[4];
    const float* att_s1   = (const float*)d_in[5];
    const float* att_d1   = (const float*)d_in[6];
    const float* b1       = (const float*)d_in[7];
    const float* W2       = (const float*)d_in[8];
    const float* att_s2   = (const float*)d_in[9];
    const float* att_d2   = (const float*)d_in[10];
    const float* b2       = (const float*)d_in[11];

    const int N  = in_sizes[2];        // 20000
    const int E  = in_sizes[1] / 2;    // 100000
    const int D  = 512, H1 = 8, C1 = 64, H2 = 4, C2 = 128, HC = 512;
    const int Etot = E + N;

    // scratch carve-out
    char* ws = (char*)d_ws; size_t off = 0;
    auto carve = [&](size_t bytes) -> void* {
        void* p = ws + off; off += (bytes + 255) & ~(size_t)255; return p;
    };
    _Float16* W1t   = (_Float16*)carve((size_t)D * HC * 2);
    _Float16* W2t   = (_Float16*)carve((size_t)HC * HC * 2);
    _Float16* bufAh = (_Float16*)carve((size_t)N * HC * 2);  // f16 A: x, then h
    float* bufA     = (float*)carve((size_t)N * HC * 4);     // xp1 then xp2
    float* bufB     = (float*)carve((size_t)N * HC * 4);     // agg1 then agg2
    float* asrc1    = (float*)carve((size_t)N * H1 * 4);
    float* adst1    = (float*)carve((size_t)N * H1 * 4);
    float* m1       = (float*)carve((size_t)N * H1 * 4);
    float* den1     = (float*)carve((size_t)N * H1 * 4);
    float* ex1      = (float*)carve((size_t)Etot * H1 * 4);
    float* asrc2    = (float*)carve((size_t)N * H2 * 4);
    float* adst2    = (float*)carve((size_t)N * H2 * 4);
    float* m2       = (float*)carve((size_t)N * H2 * 4);
    float* den2     = (float*)carve((size_t)N * H2 * 4);
    float* ex2      = (float*)carve((size_t)Etot * H2 * 4);
    float* acc      = (float*)carve(2 * 4);

    float* loss_out   = (float*)d_out;
    float* logits_out = (float*)d_out + 1;   // [N, C2]

    // 0) weights -> f16 transposed; x -> f16
    transpose_f16_k<<<gdiv((long)D * HC, TPB), TPB, 0, stream>>>(W1, W1t, D, HC);
    transpose_f16_k<<<gdiv((long)HC * HC, TPB), TPB, 0, stream>>>(W2, W2t, HC, HC);
    cvt_f16_k<<<gdiv((long)N * HC, TPB), TPB, 0, stream>>>(x, bufAh, (long)N * HC);

    // 1) GEMM1: xp1 = x @ W1  (16x64 strip per wave, 4 WMMA per k-step)
    int gemmBlocks = gdiv((long)(N / 16) * (HC / 64), 8);
    gemm_wmma_f16_k<<<gemmBlocks, TPB, 0, stream>>>(bufAh, W1t, bufA, N);

    // 2) attention scalars, layer 1
    att_score_k<<<gdiv((long)N * H1, TPB), TPB, 0, stream>>>(bufA, att_s1, asrc1, N, H1, C1);
    att_score_k<<<gdiv((long)N * H1, TPB), TPB, 0, stream>>>(bufA, att_d1, adst1, N, H1, C1);

    // 3) softmax-aggregate, layer 1
    fill_f32_k<<<gdiv((long)N * H1, TPB), TPB, 0, stream>>>(m1,  -1e30f, (long)N * H1);
    fill_f32_k<<<gdiv((long)N * H1, TPB), TPB, 0, stream>>>(den1, 0.0f,  (long)N * H1);
    fill_f32_k<<<gdiv((long)N * HC, TPB), TPB, 0, stream>>>(bufB, 0.0f,  (long)N * HC);
    edge_max_k<<<gdiv((long)Etot * H1, TPB), TPB, 0, stream>>>(ei, E, N, H1, asrc1, adst1, m1);
    edge_exp_k<<<gdiv((long)Etot * H1, TPB), TPB, 0, stream>>>(ei, E, N, H1, asrc1, adst1, m1, ex1, den1);
    agg_k<<<gdiv((long)Etot * H1 * (C1 / 8), TPB), TPB, 0, stream>>>(ei, E, N, H1, C1, ex1, den1, bufA, bufB);

    // 4) h = elu(agg + b1) -> f16 (input of GEMM2)
    elu_bias_f16_k<<<gdiv((long)N * HC, TPB), TPB, 0, stream>>>(bufB, b1, bufAh, (long)N * HC, HC);

    // 5) GEMM2: xp2 = h @ W2
    gemm_wmma_f16_k<<<gemmBlocks, TPB, 0, stream>>>(bufAh, W2t, bufA, N);

    // 6) attention scalars, layer 2
    att_score_k<<<gdiv((long)N * H2, TPB), TPB, 0, stream>>>(bufA, att_s2, asrc2, N, H2, C2);
    att_score_k<<<gdiv((long)N * H2, TPB), TPB, 0, stream>>>(bufA, att_d2, adst2, N, H2, C2);

    // 7) softmax-aggregate, layer 2 (reuse bufB as [N,H2,C2])
    fill_f32_k<<<gdiv((long)N * H2, TPB), TPB, 0, stream>>>(m2,  -1e30f, (long)N * H2);
    fill_f32_k<<<gdiv((long)N * H2, TPB), TPB, 0, stream>>>(den2, 0.0f,  (long)N * H2);
    fill_f32_k<<<gdiv((long)N * HC, TPB), TPB, 0, stream>>>(bufB, 0.0f,  (long)N * HC);
    edge_max_k<<<gdiv((long)Etot * H2, TPB), TPB, 0, stream>>>(ei, E, N, H2, asrc2, adst2, m2);
    edge_exp_k<<<gdiv((long)Etot * H2, TPB), TPB, 0, stream>>>(ei, E, N, H2, asrc2, adst2, m2, ex2, den2);
    agg_k<<<gdiv((long)Etot * H2 * (C2 / 8), TPB), TPB, 0, stream>>>(ei, E, N, H2, C2, ex2, den2, bufA, bufB);

    // 8) head mean + b2 -> logits (directly into d_out[1:])
    mean_bias_k<<<gdiv((long)N * C2, TPB), TPB, 0, stream>>>(bufB, b2, logits_out, N, H2, C2);

    // 9) masked-mean cross entropy -> d_out[0]
    fill_f32_k<<<1, TPB, 0, stream>>>(acc, 0.0f, 2);
    ce_k<<<gdiv((long)N, TPB), TPB, 0, stream>>>(logits_out, y, mask, acc, N, C2);
    finalize_k<<<1, 1, 0, stream>>>(acc, loss_out);
}
